// GlobalAttention_63015760166921
// MI455X (gfx1250) — matrix operand order repaired and checked by reading, not compiled, not run on hardware
//
#include <hip/hip_runtime.h>
#include <hip/hip_bf16.h>
#include <math.h>

#define B_ 32
#define S_ 2048
#define D_ 1024
#define TEMP_ 9.0f

typedef _Float16 v16h __attribute__((ext_vector_type(16)));
typedef _Float16 h8   __attribute__((ext_vector_type(8)));
typedef _Float16 h4   __attribute__((ext_vector_type(4)));
typedef float    v8f  __attribute__((ext_vector_type(8)));

// workspace layout (float offsets); total ~3.7 MB
#define OFF_E          0        // B*S            = 65536
#define OFF_LOGITS     65536    // B*S            = 65536
#define OFF_VMAX       131072   // S              = 2048
#define OFF_HTDOT      133120   // B              = 32
#define OFF_CPART      133632   // 8*B*D          = 262144
#define OFF_C          395776   // B*D            = 32768
#define OFF_W1H_FLOATS 428544   // D*D halves     = 524288 floats worth

// ---------------- W1 fp32 -> fp16 ----------------
__global__ void k_w1cvt(const float* __restrict__ W1, _Float16* __restrict__ W1h) {
    int i = (blockIdx.x * 256 + threadIdx.x) * 4;           // 1024 blocks * 256 * 4 = D*D
    float4 v = *(const float4*)(W1 + i);
    h4 h; h[0] = (_Float16)v.x; h[1] = (_Float16)v.y; h[2] = (_Float16)v.z; h[3] = (_Float16)v.w;
    *(h4*)(W1h + i) = h;
}

// ---------------- htdot[b] = h_t[b,:] . Wmap[D:2D] ----------------
__global__ void k_htdot(const float* __restrict__ h_t, const float* __restrict__ Wmap,
                        float* __restrict__ htdot) {
    int b = threadIdx.x;                                    // 32 threads
    const float* hp = h_t + (size_t)b * D_;
    const float* wp = Wmap + D_;
    float s = 0.f;
    for (int j = 0; j < D_; ++j) s += hp[j] * wp[j];
    htdot[b] = s;
}

// ---------------- logits[b,s] = auxi . Wm_a + htdot[b] + bmap ----------------
__global__ void k_logits(const float* __restrict__ auxi, const float* __restrict__ Wmap,
                         const float* __restrict__ bmapP, const float* __restrict__ htdot,
                         float* __restrict__ logits) {
    const int wave = threadIdx.x >> 5, lane = threadIdx.x & 31;
    const size_t row = (size_t)blockIdx.x * 8 + wave;       // 8192 blocks -> 65536 rows
    const float4* p  = (const float4*)(auxi + row * D_);
    const float4* wm = (const float4*)Wmap;                 // Wm_a = Wmap[:D]
    float s = 0.f;
    for (int j = lane; j < D_ / 4; j += 32) {
        float4 a = p[j], w = wm[j];
        s += a.x * w.x + a.y * w.y + a.z * w.z + a.w * w.w;
    }
    s += __shfl_xor(s, 16, 32);
    s += __shfl_xor(s, 8, 32);
    s += __shfl_xor(s, 4, 32);
    s += __shfl_xor(s, 2, 32);
    s += __shfl_xor(s, 1, 32);
    if (lane == 0) {
        int b = (int)(row >> 11);                           // S = 2048
        logits[row] = s + htdot[b] + bmapP[0];
    }
}

// ---------------- e[b,s] = tanh(h_s @ W1^T + b1) . W2 + b2  (WMMA) ----------------
// One WG = 64 rows of h_s staged to LDS as f16. 8 waves: wave = (msub in 0..3, nhalf in 0..1).
// Each wave: 32 n-tiles x 32 k-steps of v_wmma_f32_16x16x32_f16, fused tanh/W2 reduction.
__global__ void __launch_bounds__(256, 1) k_e(
    const float* __restrict__ h_s, const _Float16* __restrict__ W1h,
    const float* __restrict__ b1, const float* __restrict__ W2,
    const float* __restrict__ b2p, float* __restrict__ e_out)
{
    extern __shared__ _Float16 As[];                        // 64 * 1024 halves = 128 KB
    __shared__ float e_sh[128];                             // [nhalf][row 0..63]
    const int tid = threadIdx.x;
    const size_t rowBase = (size_t)blockIdx.x * 64;

    // stage 64 rows, converting fp32 -> fp16
    {
        const float4* src = (const float4*)(h_s + rowBase * D_);
        h4* dst = (h4*)As;
        for (int i = tid; i < 64 * D_ / 4; i += 256) {
            float4 v = src[i];
            h4 h; h[0] = (_Float16)v.x; h[1] = (_Float16)v.y;
            h[2] = (_Float16)v.z; h[3] = (_Float16)v.w;
            dst[i] = h;
        }
        // prefetch next block's 256KB into L2 (global_prefetch_b8)
        if (blockIdx.x + 1 < gridDim.x) {
            const char* nxt = (const char*)(h_s + (rowBase + 64) * D_);
            for (int i = tid; i < 2048; i += 256)
                __builtin_prefetch(nxt + (size_t)i * 128, 0, 1);
        }
    }
    __syncthreads();

    const int wave  = tid >> 5;
    const int lane  = tid & 31;
    const int m0    = (wave & 3) * 16;
    const int nhalf = wave >> 2;
    const bool loH  = lane < 16;
    // A fragment (16-bit 16x32 ISA layout): lanes 0-15 hold K 0..7 & 16..23 of row M=lane;
    // lanes 16-31 hold K 8..15 & 24..31 of row M=lane-16.
    const _Float16* arow = As + (size_t)(m0 + (lane & 15)) * D_ + (loH ? 0 : 8);
    // B fragment (K x N): lane = N (mod 16); lanes 0-15 -> K 0..7 & 16..23, lanes 16-31 -> K 8..15 & 24..31,
    // same K order as A, read along a W1 row (row-major over k).
    const int nB    = lane & 15;
    const int kBoff = loH ? 0 : 8;

    float ep[8];
#pragma unroll
    for (int r = 0; r < 8; ++r) ep[r] = 0.f;

    for (int nt = 0; nt < 32; ++nt) {
        const int n0 = nhalf * 512 + nt * 16;
        const _Float16* wrow = W1h + (size_t)(n0 + nB) * D_ + kBoff;
        v8f acc = {0.f, 0.f, 0.f, 0.f, 0.f, 0.f, 0.f, 0.f};
#pragma unroll 4
        for (int k0 = 0; k0 < D_; k0 += 32) {
            h8 a_lo = *(const h8*)(arow + k0);
            h8 a_hi = *(const h8*)(arow + k0 + 16);
            v16h af = __builtin_shufflevector(a_lo, a_hi,
                0, 1, 2, 3, 4, 5, 6, 7, 8, 9, 10, 11, 12, 13, 14, 15);
            h8 b_lo = *(const h8*)(wrow + k0);
            h8 b_hi = *(const h8*)(wrow + k0 + 16);
            v16h bf = __builtin_shufflevector(b_lo, b_hi,
                0, 1, 2, 3, 4, 5, 6, 7, 8, 9, 10, 11, 12, 13, 14, 15);
            acc = __builtin_amdgcn_wmma_f32_16x16x32_f16(
                false, af, false, bf, (short)0, acc, false, false);
        }
        const int ng = n0 + nB;
        const float b1n = b1[ng];
        const float w2n = W2[ng];
#pragma unroll
        for (int r = 0; r < 8; ++r)
            ep[r] += tanhf(acc[r] + b1n) * w2n;
    }

    // deterministic cross-lane reduction: component r of half-lane group = row m0+r (+8 for hi half)
#pragma unroll
    for (int r = 0; r < 8; ++r) {
        float v = ep[r];
        v += __shfl_xor(v, 1, 16);
        v += __shfl_xor(v, 2, 16);
        v += __shfl_xor(v, 4, 16);
        v += __shfl_xor(v, 8, 16);
        if ((lane & 15) == 0)
            e_sh[nhalf * 64 + m0 + r + (loH ? 0 : 8)] = v;
    }
    __syncthreads();
    if (tid < 64)
        e_out[rowBase + tid] = e_sh[tid] + e_sh[64 + tid] + b2p[0];
}

// ---------------- vmax[s] = max_b e[b,s] ----------------
__global__ void k_vmax(const float* __restrict__ e, float* __restrict__ vmax) {
    int s = blockIdx.x * 256 + threadIdx.x;                 // 8 blocks
    float m = -INFINITY;
    for (int b = 0; b < B_; ++b) m = fmaxf(m, e[(size_t)b * S_ + s]);
    vmax[s] = m;
}

// ---------------- scores, row-sum, align (one WG per batch b) ----------------
__global__ void k_align(const float* __restrict__ e, const float* __restrict__ vmax,
                        const float* __restrict__ logits, const float* __restrict__ noise,
                        float* __restrict__ alignOut) {
    const int b = blockIdx.x, tid = threadIdx.x;
    __shared__ float red[256];
    float local = 0.f;
    for (int s = tid; s < S_; s += 256) {
        const size_t i = (size_t)b * S_ + s;
        float u = noise[i];
        u = fminf(fmaxf(u, 1e-7f), 1.f - 1e-7f);
        float lgt = logf(u) - log1pf(-u);
        float z = (logits[i] + lgt) * (1.f / TEMP_);
        float G = 1.f / (1.f + expf(-z));
        float sc = expf(e[i] - vmax[s]) * G;
        alignOut[i] = sc;
        local += sc;
    }
    red[tid] = local;
    __syncthreads();
    for (int off = 128; off > 0; off >>= 1) {
        if (tid < off) red[tid] += red[tid + off];
        __syncthreads();
    }
    const float inv = 1.f / red[0];
    for (int s = tid; s < S_; s += 256)
        alignOut[(size_t)b * S_ + s] *= inv;
}

// ---------------- c partials: cpart[sc][b][d] = sum_{s in chunk sc} align*h_s ----------------
__global__ void k_cpart(const float* __restrict__ h_s, const float* __restrict__ alignv,
                        float* __restrict__ cpart) {
    const int b = blockIdx.x >> 3, sc = blockIdx.x & 7, tid = threadIdx.x; // 256 blocks
    __shared__ float al[256];
    al[tid] = alignv[(size_t)b * S_ + sc * 256 + tid];
    __syncthreads();
    const float4* hp = (const float4*)(h_s + ((size_t)b * S_ + (size_t)sc * 256) * D_);
    float4 acc = {0.f, 0.f, 0.f, 0.f};
    for (int s = 0; s < 256; ++s) {
        const float a = al[s];
        float4 h = hp[(size_t)s * (D_ / 4) + tid];
        acc.x += a * h.x; acc.y += a * h.y; acc.z += a * h.z; acc.w += a * h.w;
    }
    *(float4*)(cpart + (size_t)(sc * B_ + b) * D_ + tid * 4) = acc;
}

// ---------------- reduce 8 c-partials ----------------
__global__ void k_credux(const float* __restrict__ cpart, float* __restrict__ c) {
    int i = blockIdx.x * 256 + threadIdx.x;                 // 32 blocks, float4 index
    float4 s = {0.f, 0.f, 0.f, 0.f};
    for (int ch = 0; ch < 8; ++ch) {
        float4 v = ((const float4*)(cpart + (size_t)ch * B_ * D_))[i];
        s.x += v.x; s.y += v.y; s.z += v.z; s.w += v.w;
    }
    ((float4*)c)[i] = s;
}

// ---------------- attn_h = tanh([c, h_t] @ Wout^T) ----------------
__global__ void k_out(const float* __restrict__ c, const float* __restrict__ h_t,
                      const float* __restrict__ Wout, float* __restrict__ out) {
    const int o = blockIdx.x * 256 + threadIdx.x;           // 128 blocks -> B*D outputs
    const int b = o >> 10, dout = o & 1023;
    const float4* wc = (const float4*)(Wout + (size_t)dout * (2 * D_));
    const float4* wh = wc + D_ / 4;
    const float4* cc = (const float4*)(c + (size_t)b * D_);
    const float4* hh = (const float4*)(h_t + (size_t)b * D_);
    float acc = 0.f;
    for (int j = 0; j < D_ / 4; ++j) {
        float4 w = wc[j], x = cc[j];
        acc += w.x * x.x + w.y * x.y + w.z * x.z + w.w * x.w;
    }
    for (int j = 0; j < D_ / 4; ++j) {
        float4 w = wh[j], x = hh[j];
        acc += w.x * x.x + w.y * x.y + w.z * x.z + w.w * x.w;
    }
    out[o] = tanhf(acc);
}

extern "C" void kernel_launch(void* const* d_in, const int* in_sizes, int n_in,
                              void* d_out, int out_size, void* d_ws, size_t ws_size,
                              hipStream_t stream) {
    const float* h_t   = (const float*)d_in[0];
    const float* h_s   = (const float*)d_in[1];
    const float* auxi  = (const float*)d_in[2];
    const float* noise = (const float*)d_in[3];
    const float* Wmap  = (const float*)d_in[4];
    const float* bmap  = (const float*)d_in[5];
    const float* W1    = (const float*)d_in[6];
    const float* b1    = (const float*)d_in[7];
    const float* W2    = (const float*)d_in[8];
    const float* b2    = (const float*)d_in[9];
    const float* Wout  = (const float*)d_in[10];

    float* ws      = (float*)d_ws;
    float* e       = ws + OFF_E;
    float* logits  = ws + OFF_LOGITS;
    float* vmax    = ws + OFF_VMAX;
    float* htdot   = ws + OFF_HTDOT;
    float* cpart   = ws + OFF_CPART;
    float* cvec    = ws + OFF_C;
    _Float16* W1h  = (_Float16*)(ws + OFF_W1H_FLOATS);

    float* attn  = (float*)d_out;                 // (B, D)
    float* align = (float*)d_out + B_ * D_;       // (B, S)

    k_w1cvt <<<D_ * D_ / 1024, 256, 0, stream>>>(W1, W1h);
    k_htdot <<<1, 32, 0, stream>>>(h_t, Wmap, htdot);
    k_logits<<<B_ * S_ / 8, 256, 0, stream>>>(auxi, Wmap, bmap, htdot, logits);
    k_e     <<<B_ * S_ / 64, 256, 64 * D_ * sizeof(_Float16), stream>>>(h_s, W1h, b1, W2, b2, e);
    k_vmax  <<<S_ / 256, 256, 0, stream>>>(e, vmax);
    k_align <<<B_, 256, 0, stream>>>(e, vmax, logits, noise, align);
    k_cpart <<<B_ * 8, 256, 0, stream>>>(h_s, align, cpart);
    k_credux<<<B_ * D_ / 1024, 256, 0, stream>>>(cpart, cvec);
    k_out   <<<B_ * D_ / 256, 256, 0, stream>>>(cvec, h_t, Wout, attn);
}
